// ModulatedRoIAlignRotatedPack_82008105550114
// MI455X (gfx1250) — matrix operand order repaired and checked by reading, not compiled, not run on hardware
//
#include <hip/hip_runtime.h>
#include <math.h>

// ---------------- problem constants (from reference setup_inputs) ----------
#define NB    2          // batch
#define NC    256        // channels
#define NH    256
#define NW    256
#define NROI  1024
#define OUTD  7
#define NBIN  49         // 7*7
#define IC    12544      // 49*256  (K of GEMM1)
#define FC    1024       // hidden  (N of GEMM1, K of GEMM2)
#define KBLK  32
#define KSTEPS (IC / KBLK)   // 392
#define LDSK  40             // 32 + 8 pad (80B row stride, 16B aligned)

typedef __attribute__((ext_vector_type(16))) __bf16 v16bf;
typedef __attribute__((ext_vector_type(8)))  float  v8f;

union FragB { uint4 q[2]; v16bf v; };

static __device__ inline void split_bf16(float f, unsigned short& hi, unsigned short& lo) {
    union { __bf16 b; unsigned short u; } cv;
    __bf16 h = (__bf16)f;                 // RNE
    cv.b = h; hi = cv.u;
    float r = f - (float)h;               // exact residual in f32
    cv.b = (__bf16)r; lo = cv.u;
}

// ============================================================================
// Kernel 1: rotated RoI-Align.  One block = one (roi, bin); thread = channel.
// Writes x in GEMM-flat layout: x[r][c*49 + oy*7 + ox]  (== (n,C,7,7) flat).
// ============================================================================
__global__ __launch_bounds__(256) void roi_align_rot_kernel(
    const float* __restrict__ feat, const float* __restrict__ rois,
    float* __restrict__ xout)
{
    const int blk = blockIdx.x;
    const int r   = blk / NBIN;
    const int bin = blk - r * NBIN;
    const int oy  = bin / OUTD, ox = bin - oy * OUTD;

    const float* roi = rois + r * 6;
    const int   b  = (int)roi[0];
    const float cx = roi[1] * 0.125f;
    const float cy = roi[2] * 0.125f;
    const float rw = fmaxf(roi[3] * 0.125f, 1.0f);
    const float rh = fmaxf(roi[4] * 0.125f, 1.0f);
    const float th = roi[5];
    const float ct = cosf(th), st = sinf(th);
    const float bw = rw * (1.0f / OUTD), bh = rh * (1.0f / OUTD);

    const int c = threadIdx.x;
    const float* __restrict__ fb = feat + ((size_t)b * NC + c) * (size_t)(NH * NW);

    float acc = 0.0f;
#pragma unroll
    for (int sy = 0; sy < 2; ++sy) {
#pragma unroll
        for (int sx = 0; sx < 2; ++sx) {
            float yy = -rh * 0.5f + ((float)oy + ((float)sy + 0.5f) * 0.5f) * bh;
            float xx = -rw * 0.5f + ((float)ox + ((float)sx + 0.5f) * 0.5f) * bw;
            float xs = xx * ct - yy * st + cx;
            float ys = xx * st + yy * ct + cy;
            bool valid = (ys > -1.0f) && (ys < (float)NH) && (xs > -1.0f) && (xs < (float)NW);
            float ysc = fminf(fmaxf(ys, 0.0f), (float)(NH - 1));
            float xsc = fminf(fmaxf(xs, 0.0f), (float)(NW - 1));
            int yl = (int)floorf(ysc), xl = (int)floorf(xsc);
            int yh = min(yl + 1, NH - 1), xh = min(xl + 1, NW - 1);
            float ly = ysc - (float)yl, lx = xsc - (float)xl;
            float hy = 1.0f - ly, hx = 1.0f - lx;
            float v11 = fb[yl * NW + xl], v12 = fb[yl * NW + xh];
            float v21 = fb[yh * NW + xl], v22 = fb[yh * NW + xh];
            float v = hy * hx * v11 + hy * lx * v12 + ly * hx * v21 + ly * lx * v22;
            acc += valid ? v : 0.0f;
        }
    }
    xout[(size_t)r * IC + c * NBIN + bin] = acc * 0.25f;
}

// ============================================================================
// Kernel 2: GEMM1  hid = relu(x @ w1 + b1)   M=1024 K=12544 N=1024
// 64x64 tile / 128 threads (4 waves).  f32 -> (hi,lo) bf16 split, 3 WMMAs
// per accumulator tile per 32-wide k-step: hi*hi + hi*lo + lo*hi.
// ============================================================================
__global__ __launch_bounds__(128) void gemm1_wmma_kernel(
    const float* __restrict__ A,   // x  [1024][12544]
    const float* __restrict__ Bw,  // w1 [12544][1024]
    const float* __restrict__ b1,
    float* __restrict__ hid)       // [1024][1024]
{
    __shared__ unsigned short aHi[64 * LDSK];
    __shared__ unsigned short aLo[64 * LDSK];
    __shared__ unsigned short bHi[64 * LDSK];   // stored transposed: [n][k]
    __shared__ unsigned short bLo[64 * LDSK];

    const int tid  = threadIdx.x;
    const int bm   = blockIdx.x & 15;
    const int bn   = blockIdx.x >> 4;
    const int m0   = bm * 64, n0 = bn * 64;
    const int wave = tid >> 5, lane = tid & 31;
    const int lr   = lane & 15, hb = lane >> 4;
    const int mOff = (wave & 1) * 32;
    const int nOff = (wave >> 1) * 32;

    v8f acc[2][2];
#pragma unroll
    for (int mi = 0; mi < 2; ++mi)
#pragma unroll
        for (int ni = 0; ni < 2; ++ni)
            acc[mi][ni] = (v8f)0.0f;

    for (int kt = 0; kt < KSTEPS; ++kt) {
        const int k0 = kt * KBLK;
        // ---- stage A tile (64 rows x 32 k), split into hi/lo bf16 ----
#pragma unroll
        for (int p = 0; p < 4; ++p) {
            int q   = tid + p * 128;
            int row = q >> 3;
            int kq  = (q & 7) << 2;
            const float4 v = *(const float4*)(A + (size_t)(m0 + row) * IC + k0 + kq);
            int base = row * LDSK + kq;
            unsigned short h, l;
            split_bf16(v.x, h, l); aHi[base + 0] = h; aLo[base + 0] = l;
            split_bf16(v.y, h, l); aHi[base + 1] = h; aLo[base + 1] = l;
            split_bf16(v.z, h, l); aHi[base + 2] = h; aLo[base + 2] = l;
            split_bf16(v.w, h, l); aHi[base + 3] = h; aLo[base + 3] = l;
        }
        // ---- stage B tile (32 k x 64 n) transposed into LDS [n][k] ----
#pragma unroll
        for (int p = 0; p < 4; ++p) {
            int q  = tid + p * 128;
            int kr = q >> 4;
            int nq = (q & 15) << 2;
            const float4 v = *(const float4*)(Bw + (size_t)(k0 + kr) * FC + n0 + nq);
            unsigned short h, l;
            split_bf16(v.x, h, l); bHi[(nq + 0) * LDSK + kr] = h; bLo[(nq + 0) * LDSK + kr] = l;
            split_bf16(v.y, h, l); bHi[(nq + 1) * LDSK + kr] = h; bLo[(nq + 1) * LDSK + kr] = l;
            split_bf16(v.z, h, l); bHi[(nq + 2) * LDSK + kr] = h; bLo[(nq + 2) * LDSK + kr] = l;
            split_bf16(v.w, h, l); bHi[(nq + 3) * LDSK + kr] = h; bLo[(nq + 3) * LDSK + kr] = l;
        }
        __syncthreads();

        // ---- build fragments per ISA 7.12.2 layouts ----
        // A 16x32 bf16: lane lr = M row; elem i -> K = i + 8*(i>=8) + 8*hb
        FragB aFH[2], aFL[2], bFH[2], bFL[2];
#pragma unroll
        for (int mi = 0; mi < 2; ++mi) {
            int row = (mOff + mi * 16 + lr) * LDSK;
            aFH[mi].q[0] = *(const uint4*)&aHi[row + hb * 8];
            aFH[mi].q[1] = *(const uint4*)&aHi[row + 16 + hb * 8];
            aFL[mi].q[0] = *(const uint4*)&aLo[row + hb * 8];
            aFL[mi].q[1] = *(const uint4*)&aLo[row + 16 + hb * 8];
        }
        // B 32x16 bf16: lane lr = N col; elem i -> K = i + 8*(i>=8) + 8*hb
#pragma unroll
        for (int ni = 0; ni < 2; ++ni) {
            int col = (nOff + ni * 16 + lr) * LDSK;
            bFH[ni].q[0] = *(const uint4*)&bHi[col + hb * 8];
            bFH[ni].q[1] = *(const uint4*)&bHi[col + 16 + hb * 8];
            bFL[ni].q[0] = *(const uint4*)&bLo[col + hb * 8];
            bFL[ni].q[1] = *(const uint4*)&bLo[col + 16 + hb * 8];
        }

        // ---- 12 WMMAs: 4 independent 3-deep chains ----
#pragma unroll
        for (int mi = 0; mi < 2; ++mi)
#pragma unroll
            for (int ni = 0; ni < 2; ++ni) {
                acc[mi][ni] = __builtin_amdgcn_wmma_f32_16x16x32_bf16(
                    false, aFH[mi].v, false, bFH[ni].v, (short)0, acc[mi][ni], false, false);
                acc[mi][ni] = __builtin_amdgcn_wmma_f32_16x16x32_bf16(
                    false, aFH[mi].v, false, bFL[ni].v, (short)0, acc[mi][ni], false, false);
                acc[mi][ni] = __builtin_amdgcn_wmma_f32_16x16x32_bf16(
                    false, aFL[mi].v, false, bFH[ni].v, (short)0, acc[mi][ni], false, false);
            }
        __syncthreads();
    }

    // ---- epilogue: bias + relu.  C/D layout: VGPR r -> M = r + 8*hb, N = lr.
#pragma unroll
    for (int mi = 0; mi < 2; ++mi)
#pragma unroll
        for (int ni = 0; ni < 2; ++ni) {
            int n = n0 + nOff + ni * 16 + lr;
            float bias = b1[n];
#pragma unroll
            for (int rr = 0; rr < 8; ++rr) {
                int m = m0 + mOff + mi * 16 + hb * 8 + rr;
                float v = acc[mi][ni][rr] + bias;
                hid[(size_t)m * FC + n] = fmaxf(v, 0.0f);
            }
        }
}

// ============================================================================
// Kernel 3: GEMM2 + sigmoid.  One wave per (roi, out) element; K=1024.
// ============================================================================
__global__ __launch_bounds__(256) void gemm2_mask_kernel(
    const float* __restrict__ hid, const float* __restrict__ w2,
    const float* __restrict__ b2, float* __restrict__ mask)
{
    const int g = blockIdx.x * 8 + (threadIdx.x >> 5);
    const int lane = threadIdx.x & 31;
    if (g >= NROI * NBIN) return;
    const int r = g / NBIN, o = g - r * NBIN;
    const float* __restrict__ hrow = hid + (size_t)r * FC;
    float s = 0.0f;
    for (int k = lane; k < FC; k += 32)
        s = fmaf(hrow[k], w2[k * NBIN + o], s);
#pragma unroll
    for (int off = 16; off; off >>= 1)
        s += __shfl_xor(s, off, 32);
    if (lane == 0)
        mask[g] = 1.0f / (1.0f + expf(-(s + b2[o])));
}

// ============================================================================
// Kernel 4: out = x * mask (mask broadcast over channels).
// ============================================================================
__global__ __launch_bounds__(256) void modulate_kernel(
    const float* __restrict__ x, const float* __restrict__ mask,
    float* __restrict__ out)
{
    const size_t i = (size_t)blockIdx.x * 256 + threadIdx.x;  // exactly 1024*12544
    const int r   = (int)(i / IC);
    const int bin = (int)(i % NBIN);                           // IC % 49 == 0
    out[i] = x[i] * mask[r * NBIN + bin];
}

// ============================================================================
extern "C" void kernel_launch(void* const* d_in, const int* in_sizes, int n_in,
                              void* d_out, int out_size, void* d_ws, size_t ws_size,
                              hipStream_t stream)
{
    const float* feat = (const float*)d_in[0];
    const float* rois = (const float*)d_in[1];
    const float* w1   = (const float*)d_in[2];
    const float* b1   = (const float*)d_in[3];
    const float* w2   = (const float*)d_in[4];
    const float* b2   = (const float*)d_in[5];
    float* out = (float*)d_out;

    // workspace layout (f32): x[1024*12544] | hid[1024*1024] | mask[1024*49]
    float* xf   = (float*)d_ws;
    float* hid  = xf + (size_t)NROI * IC;
    float* mask = hid + (size_t)NROI * FC;

    roi_align_rot_kernel<<<NROI * NBIN, 256, 0, stream>>>(feat, rois, xf);
    gemm1_wmma_kernel<<<(NROI / 64) * (FC / 64), 128, 0, stream>>>(xf, w1, b1, hid);
    gemm2_mask_kernel<<<(NROI * NBIN) / 8, 256, 0, stream>>>(hid, w2, b2, mask);
    modulate_kernel<<<(NROI * IC) / 256, 256, 0, stream>>>(xf, mask, out);
}
